// MambaBlock_20641612825423
// MI455X (gfx1250) — matrix operand order repaired and checked by reading, not compiled, not run on hardware
//
#include <hip/hip_runtime.h>
#include <hip/hip_bf16.h>

// ---------------------------------------------------------------------------
// Mamba block forward for MI455X (gfx1250, wave32, WMMA + TDM).
// GEMMs: v_wmma_f32_16x16x32_bf16, fragment-ready LDS tiles (every fragment
// is one contiguous 32B per-lane read -> 2x ds_load_b128), double-buffered
// LDS, register-staged global->LDS pipeline, native f32->bf16 converts.
// Scan: register-resident 16-state recurrence per (b, channel); B/C selector
// tiles staged by the Tensor Data Mover (tensor_load_to_lds + s_wait_tensorcnt).
// Sizes fixed: B=2, L=1024, D_MODEL=1024, D_INNER=2048, D_STATE=16, DT_RANK=64.
// ---------------------------------------------------------------------------

typedef __attribute__((ext_vector_type(16))) __bf16 v16bf;
typedef __attribute__((ext_vector_type(2)))  __bf16 v2bf;
typedef __attribute__((ext_vector_type(8)))  float  v8f;
typedef __attribute__((ext_vector_type(4)))  unsigned int v4u;
typedef __attribute__((ext_vector_type(8)))  int    v8i;
typedef __attribute__((ext_vector_type(4)))  int    v4i;

#define BATCH    2
#define SEQ_L    1024
#define DMODEL   1024
#define DINNER   2048
#define DSTATE   16
#define DCONV    4
#define DTRANK   64
#define NROWS    (BATCH * SEQ_L)       // GEMM M = 2048

// 2x f32 -> packed bf16 pair. Prefer the packed-convert builtin; else rely on
// native fptrunc(f32->bf16) selection (v_cvt_bf16_f32 on bf16-native gfx1250).
__device__ __forceinline__ unsigned pack2bf(float a, float b) {
#if defined(__gfx1250__) && __has_builtin(__builtin_amdgcn_cvt_pk_bf16_f32)
  v2bf p = __builtin_amdgcn_cvt_pk_bf16_f32(a, b);
#else
  v2bf p;
  p[0] = (__bf16)a;
  p[1] = (__bf16)b;
#endif
  union { v2bf v; unsigned u; } o; o.v = p;
  return o.u;
}
__device__ __forceinline__ float softplus_f(float x) {
  return (x > 20.f) ? x : log1pf(__expf(x));
}
__device__ __forceinline__ float silu_f(float x) {
  return x / (1.f + __expf(-x));
}

// ---------------------------------------------------------------------------
// GEMM: C[M,N] = act( A[M,K] * W[N,K]^T + bias[N] )
// 256 threads = 8 waves as 4(M) x 2(N); wave tile 32x32 (4 accumulators);
// block tile 128x64; BK=64 -> 8 v_wmma per K iteration.
//
// Fragment-ready LDS layout (wave32, ISA 7.12.2):
//   A frag element i of lane (h*16+r): K = (i<8) ? h*8+i : 16+h*8+(i-8), M=r
//   B frag element i of lane (h*16+r): K = (i<8) ? h*8+i : 16+h*8+(i-8), N=r
// lAf[ks][mt][lane][16], lBf[ks][nt][lane][16]  (bf16; 32B per lane entry)
// ---------------------------------------------------------------------------
#define BM 128
#define BN 64
#define BK 64
#define LAF_SZ (2 * 8 * 32 * 16)   // 8192 bf16 = 16KB
#define LBF_SZ (2 * 4 * 32 * 16)   // 4096 bf16 =  8KB

#define GEMM_LOAD_TILE(K0)                                                     \
  do {                                                                         \
    _Pragma("unroll")                                                          \
    for (int i = 0; i < 8; ++i) {                                              \
      int li = i * 256 + tid, row = li >> 4, c4 = (li & 15) << 2;              \
      ra[i] = *(const float4*)(A + (size_t)(m0 + row) * lda + (K0) + c4);      \
    }                                                                          \
    _Pragma("unroll")                                                          \
    for (int i = 0; i < 4; ++i) {                                              \
      int li = i * 256 + tid, n = li >> 4, c4 = (li & 15) << 2;                \
      int gn = n0 + n;                                                         \
      rb[i] = (gn < N) ? *(const float4*)(W + (size_t)gn * ldw + (K0) + c4)    \
                       : make_float4(0.f, 0.f, 0.f, 0.f);                      \
    }                                                                          \
  } while (0)

#define GEMM_STORE_TILE(BUF)                                                   \
  do {                                                                         \
    _Pragma("unroll")                                                          \
    for (int i = 0; i < 8; ++i) {                                              \
      int li = i * 256 + tid, row = li >> 4, c4 = (li & 15) << 2;              \
      int mt = row >> 4, rr = row & 15;                                        \
      int ks = c4 >> 5, kk = c4 & 31;                                          \
      int hh = (kk >> 3) & 1, ln = hh * 16 + rr;                               \
      int i0 = (kk & 7) + ((kk >> 4) << 3);                                    \
      *(uint2*)&lAf[BUF][(((ks << 3) + mt) * 32 + ln) * 16 + i0] =             \
          make_uint2(pack2bf(ra[i].x, ra[i].y), pack2bf(ra[i].z, ra[i].w));    \
    }                                                                          \
    _Pragma("unroll")                                                          \
    for (int i = 0; i < 4; ++i) {                                              \
      int li = i * 256 + tid, n = li >> 4, c4 = (li & 15) << 2;                \
      int nt = n >> 4, rr = n & 15;                                            \
      int ks = c4 >> 5, kk = c4 & 31;                                          \
      int hh = (kk >> 3) & 1, ln = hh * 16 + rr;                               \
      int i0 = (kk & 7) + ((kk >> 4) << 3);                                    \
      *(uint2*)&lBf[BUF][(((ks << 2) + nt) * 32 + ln) * 16 + i0] =             \
          make_uint2(pack2bf(rb[i].x, rb[i].y), pack2bf(rb[i].z, rb[i].w));    \
    }                                                                          \
  } while (0)

__global__ void __launch_bounds__(256)
gemm_wmma_bf16(const float* __restrict__ A, const float* __restrict__ W,
               float* __restrict__ C, const float* __restrict__ bias,
               int M, int N, int K, int lda, int ldw, int ldc, int act)
{
  __shared__ __attribute__((aligned(32))) __bf16 lAf[2][LAF_SZ];
  __shared__ __attribute__((aligned(32))) __bf16 lBf[2][LBF_SZ];

  const int tid   = threadIdx.x;
  const int lane  = tid & 31;
  const int wave  = tid >> 5;          // 0..7
  const int waveM = wave >> 1;         // 0..3 (32 rows each)
  const int waveN = wave & 1;          // 0..1 (32 cols each)
  const int h     = lane >> 4;
  const int r     = lane & 15;

  const int m0 = blockIdx.x * BM;
  const int n0 = blockIdx.y * BN;

  v8f acc[2][2] = {};
  float4 ra[8], rb[4];

  GEMM_LOAD_TILE(0);
  GEMM_STORE_TILE(0);
  __syncthreads();

  int buf = 0;
  for (int k0 = 0; k0 < K; k0 += BK) {
    const bool more = (k0 + BK) < K;
    if (more) GEMM_LOAD_TILE(k0 + BK);   // overlap global loads with WMMA

    #pragma unroll
    for (int ks = 0; ks < 2; ++ks) {
      v16bf a0 = *(const v16bf*)&lAf[buf][(((ks << 3) + (waveM * 2 + 0)) * 32 + lane) * 16];
      v16bf a1 = *(const v16bf*)&lAf[buf][(((ks << 3) + (waveM * 2 + 1)) * 32 + lane) * 16];
      v16bf b0 = *(const v16bf*)&lBf[buf][(((ks << 2) + (waveN * 2 + 0)) * 32 + lane) * 16];
      v16bf b1 = *(const v16bf*)&lBf[buf][(((ks << 2) + (waveN * 2 + 1)) * 32 + lane) * 16];
      acc[0][0] = __builtin_amdgcn_wmma_f32_16x16x32_bf16(false, a0, false, b0, (short)0, acc[0][0], false, false);
      acc[0][1] = __builtin_amdgcn_wmma_f32_16x16x32_bf16(false, a0, false, b1, (short)0, acc[0][1], false, false);
      acc[1][0] = __builtin_amdgcn_wmma_f32_16x16x32_bf16(false, a1, false, b0, (short)0, acc[1][0], false, false);
      acc[1][1] = __builtin_amdgcn_wmma_f32_16x16x32_bf16(false, a1, false, b1, (short)0, acc[1][1], false, false);
    }

    if (more) {
      GEMM_STORE_TILE(buf ^ 1);
      __syncthreads();
      buf ^= 1;
    }
  }

  // epilogue: C/D layout row = h*8 + vgpr_idx, col = lane & 15
  #pragma unroll
  for (int m = 0; m < 2; ++m) {
    #pragma unroll
    for (int n = 0; n < 2; ++n) {
      int gc = n0 + waveN * 32 + n * 16 + r;
      if (gc >= N) continue;
      float bv = bias ? bias[gc] : 0.f;
      #pragma unroll
      for (int i = 0; i < 8; ++i) {
        int gr = m0 + waveM * 32 + m * 16 + h * 8 + i;
        float v = acc[m][n][i] + bv;
        if (act == 1) v = softplus_f(v);
        C[(size_t)gr * ldc + gc] = v;
      }
    }
  }
}

// ---------------------------------------------------------------------------
// Causal depthwise conv (4 taps) + bias + SiLU.
// ---------------------------------------------------------------------------
__global__ void __launch_bounds__(256)
conv_silu_kernel(const float* __restrict__ xz, const float* __restrict__ conv_w,
                 const float* __restrict__ conv_b, float* __restrict__ xconv)
{
  int idx = blockIdx.x * 256 + threadIdx.x;          // over B*L*DINNER
  if (idx >= BATCH * SEQ_L * DINNER) return;
  int d   = idx & (DINNER - 1);
  int bl  = idx / DINNER;                            // b*L + l
  int l   = bl & (SEQ_L - 1);
  float acc = conv_b[d];
  #pragma unroll
  for (int j = 0; j < DCONV; ++j) {
    int lt = l - (DCONV - 1) + j;
    if (lt >= 0)
      acc += xz[(size_t)(bl + (lt - l)) * (2 * DINNER) + d] * conv_w[d * DCONV + j];
  }
  xconv[idx] = silu_f(acc);
}

// ---------------------------------------------------------------------------
// Selective scan. Thread per (b, d); 16 states in registers.
// B/C selectors (dbc cols 64..95, row stride 96) for 32 timesteps are a 2D
// strided tile -> staged into LDS by the Tensor Data Mover (wave 0 issues,
// s_wait_tensorcnt, then workgroup barrier). Fallback: cooperative loads.
// ---------------------------------------------------------------------------
#define TCHUNK 32

__global__ void __launch_bounds__(256)
scan_kernel(const float* __restrict__ xconv, const float* __restrict__ delta,
            const float* __restrict__ dbc,   const float* __restrict__ xz,
            const float* __restrict__ A_log, const float* __restrict__ D_param,
            float* __restrict__ ygated)
{
  __shared__ __attribute__((aligned(16))) float sBC[TCHUNK][32]; // [t][B(16)|C(16)]

  const int tid = threadIdx.x;
  const int b   = blockIdx.y;
  const int d   = blockIdx.x * 256 + tid;

  float Ad[DSTATE];
  #pragma unroll
  for (int s = 0; s < DSTATE; ++s) Ad[s] = -__expf(A_log[d * DSTATE + s]);
  const float Dp = D_param[d];

  float hst[DSTATE];
  #pragma unroll
  for (int s = 0; s < DSTATE; ++s) hst[s] = 0.f;

  for (int t0 = 0; t0 < SEQ_L; t0 += TCHUNK) {
#if defined(__gfx1250__) && __has_builtin(__builtin_amdgcn_tensor_load_to_lds)
    if (tid < 32) {  // wave 0 drives the TDM (EXEC ignored; per-wave op)
      unsigned lds_base = (unsigned)(size_t)(void*)&sBC[0][0];
      unsigned long long ga =
          (unsigned long long)(size_t)(const void*)(dbc + (size_t)(b * SEQ_L + t0) * 96 + 64);
      // D# group 0: count=1 | lds_addr | global_addr[56:0] | type=2
      v4u g0;
      g0[0] = 1u;
      g0[1] = lds_base;
      g0[2] = (unsigned)(ga & 0xFFFFFFFFu);
      g0[3] = (unsigned)(ga >> 32) | (2u << 30);
      // D# group 1: data_size=4B, tensor 32x32 (stride 96), tile 32x32
      v8i g1;
      g1[0] = (int)(2u << 16);          // data_size code 2 (4 bytes)
      g1[1] = (int)(32u << 16);         // tensor_dim0 = 32 (low16 at bit48)
      g1[2] = (int)(32u << 16);         // tensor_dim0 hi=0 | tensor_dim1=32
      g1[3] = (int)(32u << 16);         // tensor_dim1 hi=0 | tile_dim0=32
      g1[4] = (int)32;                  // tile_dim1=32 | tile_dim2=0
      g1[5] = (int)96;                  // tensor_dim0_stride = 96 (low32)
      g1[6] = 0;                        // stride0 hi | tensor_dim1_stride lo
      g1[7] = 0;
      v4i g2 = {0, 0, 0, 0};
      v4i g3 = {0, 0, 0, 0};
#if __clang_major__ >= 23
      v8i gz = {0, 0, 0, 0, 0, 0, 0, 0};
      __builtin_amdgcn_tensor_load_to_lds(g0, g1, g2, g3, gz, 0);
#else
      __builtin_amdgcn_tensor_load_to_lds(g0, g1, g2, g3, 0);
#endif
      __builtin_amdgcn_s_wait_tensorcnt(0);
    }
    __syncthreads();
#else
    #pragma unroll
    for (int i = 0; i < (TCHUNK * 32) / 256; ++i) {
      int li = i * 256 + tid;
      sBC[li >> 5][li & 31] = dbc[(size_t)(b * SEQ_L + t0 + (li >> 5)) * 96 + 64 + (li & 31)];
    }
    __syncthreads();
#endif

    if (t0 + TCHUNK < SEQ_L) {  // global_prefetch_b8 for next chunk
      __builtin_prefetch(&delta[(size_t)(b * SEQ_L + t0 + TCHUNK) * DINNER + d], 0, 1);
      __builtin_prefetch(&xconv[(size_t)(b * SEQ_L + t0 + TCHUNK) * DINNER + d], 0, 1);
    }

    for (int tt = 0; tt < TCHUNK; ++tt) {
      const size_t bl = (size_t)(b * SEQ_L + t0 + tt);
      const float dl  = delta[bl * DINNER + d];
      const float xv  = xconv[bl * DINNER + d];
      const float zv  = xz[bl * (2 * DINNER) + DINNER + d];
      const float dx  = dl * xv;
      float y = 0.f;
      #pragma unroll
      for (int s = 0; s < DSTATE; ++s) {
        float dA = __expf(dl * Ad[s]);
        hst[s] = dA * hst[s] + dx * sBC[tt][s];
        y += hst[s] * sBC[tt][16 + s];
      }
      y += Dp * xv;
      ygated[bl * DINNER + d] = y * silu_f(zv);
    }
    __syncthreads();
  }
}

// ---------------------------------------------------------------------------
// Orchestration. Workspace (floats):
//   xz(8,388,608) | xconv(4,194,304) | dbc(196,608) | delta(4,194,304)
//   | ygated(4,194,304)
// ---------------------------------------------------------------------------
extern "C" void kernel_launch(void* const* d_in, const int* in_sizes, int n_in,
                              void* d_out, int out_size, void* d_ws, size_t ws_size,
                              hipStream_t stream) {
  const float* x       = (const float*)d_in[0];
  const float* W_in    = (const float*)d_in[1];
  const float* conv_w  = (const float*)d_in[2];
  const float* conv_b  = (const float*)d_in[3];
  const float* W_x     = (const float*)d_in[4];
  const float* W_dt    = (const float*)d_in[5];
  const float* b_dt    = (const float*)d_in[6];
  const float* A_log   = (const float*)d_in[7];
  const float* D_param = (const float*)d_in[8];
  const float* W_out   = (const float*)d_in[9];
  float* out = (float*)d_out;

  float* xz     = (float*)d_ws;
  float* xconv  = xz     + (size_t)NROWS * 2 * DINNER;
  float* dbc    = xconv  + (size_t)NROWS * DINNER;
  float* delta  = dbc    + (size_t)NROWS * 96;
  float* ygated = delta  + (size_t)NROWS * DINNER;

  dim3 blk(256);

  // 1) in-proj: xz = x @ W_in^T           (2048 x 4096 x 1024)
  gemm_wmma_bf16<<<dim3(NROWS / BM, (2 * DINNER) / BN), blk, 0, stream>>>(
      x, W_in, xz, nullptr, NROWS, 2 * DINNER, DMODEL, DMODEL, DMODEL, 2 * DINNER, 0);

  // 2) causal depthwise conv + SiLU
  conv_silu_kernel<<<dim3((BATCH * SEQ_L * DINNER) / 256), blk, 0, stream>>>(
      xz, conv_w, conv_b, xconv);

  // 3) x-proj: dbc = xconv @ W_x^T        (2048 x 96 x 2048)
  gemm_wmma_bf16<<<dim3(NROWS / BM, (96 + BN - 1) / BN), blk, 0, stream>>>(
      xconv, W_x, dbc, nullptr, NROWS, 96, DINNER, DINNER, DINNER, 96, 0);

  // 4) dt-proj: delta = softplus(dt @ W_dt^T + b_dt)   (2048 x 2048 x 64)
  gemm_wmma_bf16<<<dim3(NROWS / BM, DINNER / BN), blk, 0, stream>>>(
      dbc, W_dt, delta, b_dt, NROWS, DINNER, DTRANK, 96, DTRANK, DINNER, 1);

  // 5) selective scan + D-skip + SiLU(z) gating
  scan_kernel<<<dim3(DINNER / 256, BATCH), blk, 0, stream>>>(
      xconv, delta, dbc, xz, A_log, D_param, ygated);

  // 6) out-proj: out = ygated @ W_out^T   (2048 x 1024 x 2048)
  gemm_wmma_bf16<<<dim3(NROWS / BM, DMODEL / BN), blk, 0, stream>>>(
      ygated, W_out, out, nullptr, NROWS, DMODEL, DINNER, DINNER, DINNER, DMODEL, 0);
}
